// HGCNLayer_40879498723408
// MI455X (gfx1250) — matrix operand order repaired and checked by reading, not compiled, not run on hardware
//
#include <hip/hip_runtime.h>
#include <stdint.h>

// ---------------------------------------------------------------------------
// HAN layer for MI455X (gfx1250, wave32).
//   Phase 1: xw[p] = x @ Ws[p]        -- bf16 WMMA (v_wmma_f32_16x16x32_bf16)
//   Phase 2: degree + rsqrt norm      -- f32 atomics (L2-resident, 800 KB)
//   Phase 3: z[p] = scatter-add       -- wave-per-edge, 16B gathers, f32 atomics
//   Phase 4: semantic attention       -- VALU, LDS-staged z, shfl reduction
// ---------------------------------------------------------------------------

typedef __attribute__((ext_vector_type(16))) __bf16 v16bf;
typedef __attribute__((ext_vector_type(8)))  float  v8f;

#define NPATH 4
#define DDIM  256     // D_IN == D_OUT
#define DHID  128
#define KTILES 8      // 256 / 32
#define NTILES 16     // 256 / 16

__device__ __forceinline__ uint16_t f2bf(float f) {
  // round-to-nearest-even f32 -> bf16
  uint32_t u = __float_as_uint(f);
  u += 0x7FFFu + ((u >> 16) & 1u);
  return (uint16_t)(u >> 16);
}

// --- x (f32) -> xh (bf16) ---------------------------------------------------
__global__ void convert_x(const float* __restrict__ x, uint16_t* __restrict__ xh, int tot) {
  int i = blockIdx.x * blockDim.x + threadIdx.x;
  if (i < tot) xh[i] = f2bf(x[i]);
}

// --- pack Ws into per-lane WMMA B fragments ---------------------------------
// wpack[((p*KTILES+kt)*NTILES+nt)*32 + lane] = 8 dwords (16 bf16):
//   koff = lane>=16 ? 16 : 0, col = nt*16 + (lane&15)
//   dword v = { W[kt*32+koff+2v][col], W[kt*32+koff+2v+1][col] }
__global__ void pack_w(const float* __restrict__ Ws, uint16_t* __restrict__ wpack) {
  int tid = blockIdx.x * blockDim.x + threadIdx.x;
  if (tid >= NPATH * KTILES * NTILES * 32) return;
  int lane = tid & 31;
  int frag = tid >> 5;
  int nt = frag % NTILES;
  int kt = (frag / NTILES) % KTILES;
  int p  = frag / (NTILES * KTILES);
  int koff = (lane >> 4) ? 16 : 0;
  int col  = nt * 16 + (lane & 15);
  const float* W = Ws + (size_t)p * DDIM * DDIM;
  uint16_t* o = wpack + (size_t)tid * 16;
  #pragma unroll
  for (int v = 0; v < 8; ++v) {
    int k = kt * 32 + koff + 2 * v;
    o[2 * v]     = f2bf(W[(size_t)k * DDIM + col]);
    o[2 * v + 1] = f2bf(W[(size_t)(k + 1) * DDIM + col]);
  }
}

// --- GEMM: xw[p] = x @ Ws[p], one wave -> 16x64 output strip -----------------
__global__ __launch_bounds__(128)
void gemm_xw(const uint16_t* __restrict__ xh, const uint16_t* __restrict__ wpack,
             float* __restrict__ xw, int n, int mt_total) {
  int wid  = (blockIdx.x * blockDim.x + threadIdx.x) >> 5;
  int lane = threadIdx.x & 31;
  int wavesPerPath = mt_total * 4;
  int p = wid / wavesPerPath;
  if (p >= NPATH) return;
  int rem = wid % wavesPerPath;
  int mt = rem >> 2;   // 16-row tile index
  int ng = rem & 3;    // 64-col group

  int half = lane >> 4;
  int l15  = lane & 15;

  int arow = mt * 16 + l15;
  if (arow >= n) arow = n - 1;                   // n % 16 == 0 in practice
  const uint16_t* arowp = xh + (size_t)arow * DDIM;

  v8f zero = {0.f, 0.f, 0.f, 0.f, 0.f, 0.f, 0.f, 0.f};
  v8f acc[4] = {zero, zero, zero, zero};

  for (int kt = 0; kt < KTILES; ++kt) {
    // A fragment: lanes 0-15 hold K in [0,16); lanes 16-31 K in [16,32)
    int colbase = kt * 32 + half * 16;
    union { v16bf v; uint4 q[2]; } a;
    a.q[0] = *reinterpret_cast<const uint4*>(arowp + colbase);
    a.q[1] = *reinterpret_cast<const uint4*>(arowp + colbase + 8);
    #pragma unroll
    for (int t = 0; t < 4; ++t) {
      int nt = ng * 4 + t;
      v16bf b = *reinterpret_cast<const v16bf*>(
          wpack + (((size_t)(p * KTILES + kt) * NTILES + nt) * 32 + lane) * 16);
      acc[t] = __builtin_amdgcn_wmma_f32_16x16x32_bf16(
          false, a.v, false, b, (short)0, acc[t], false, false);
    }
  }

  // C layout: VGPR r -> row r + 8*half, col = l15
  int rowbase = mt * 16 + half * 8;
  #pragma unroll
  for (int t = 0; t < 4; ++t) {
    int col = (ng * 4 + t) * 16 + l15;
    #pragma unroll
    for (int r = 0; r < 8; ++r) {
      int row = rowbase + r;
      if (row < n)
        xw[((size_t)p * n + row) * DDIM + col] = acc[t][r];
    }
  }
}

// --- degree (in-degree incl. self loop) -------------------------------------
__global__ void deg_init(float* __restrict__ deg, int tot) {
  int i = blockIdx.x * blockDim.x + threadIdx.x;
  if (i < tot) deg[i] = 1.0f;                    // self loop
}

__global__ void deg_count(const int* __restrict__ ei, float* __restrict__ deg,
                          int n, int nE) {
  long long i = (long long)blockIdx.x * blockDim.x + threadIdx.x;
  if (i >= (long long)NPATH * nE) return;
  int p = (int)(i / nE), e = (int)(i % nE);
  int dst = ei[((size_t)p * 2 + 1) * nE + e];
  atomicAdd(&deg[(size_t)p * n + dst], 1.0f);
}

__global__ void deg_fin(const float* __restrict__ deg, float* __restrict__ dinv, int tot) {
  int i = blockIdx.x * blockDim.x + threadIdx.x;
  if (i < tot) dinv[i] = (deg[i] > 0.f) ? rsqrtf(deg[i]) : 0.f;
}

// --- z init: self-loop message + bias ---------------------------------------
__global__ void z_init(const float* __restrict__ xw, const float* __restrict__ dinv,
                       const float* __restrict__ bs, float* __restrict__ z, int n) {
  long long i = (long long)blockIdx.x * blockDim.x + threadIdx.x;
  long long tot = (long long)NPATH * n * DDIM;
  if (i >= tot) return;
  int d = (int)(i % DDIM);
  long long nd = i / DDIM;
  int node = (int)(nd % n);
  int p = (int)(nd / n);
  float di = dinv[(size_t)p * n + node];
  z[i] = xw[i] * di * di + bs[p * DDIM + d];
}

// --- edge scatter: one wave per edge, 32 lanes x 8 floats = full row --------
__global__ __launch_bounds__(256)
void edge_scatter(const int* __restrict__ ei, const float* __restrict__ xw,
                  const float* __restrict__ dinv, float* __restrict__ z,
                  int n, int nE) {
  long long gw = ((long long)blockIdx.x * blockDim.x + threadIdx.x) >> 5;
  int lane = threadIdx.x & 31;
  if (gw >= (long long)NPATH * nE) return;
  int p = (int)(gw / nE), e = (int)(gw % nE);
  const int* base = ei + (size_t)p * 2 * nE;
  int src = base[e];
  int dst = base[nE + e];
  float nrm = dinv[(size_t)p * n + src] * dinv[(size_t)p * n + dst];
  const float4* xs = reinterpret_cast<const float4*>(xw + ((size_t)p * n + src) * DDIM) + lane * 2;
  float* zd = z + ((size_t)p * n + dst) * DDIM + lane * 8;
  float4 v0 = xs[0];
  float4 v1 = xs[1];
  atomicAdd(zd + 0, v0.x * nrm); atomicAdd(zd + 1, v0.y * nrm);
  atomicAdd(zd + 2, v0.z * nrm); atomicAdd(zd + 3, v0.w * nrm);
  atomicAdd(zd + 4, v1.x * nrm); atomicAdd(zd + 5, v1.y * nrm);
  atomicAdd(zd + 6, v1.z * nrm); atomicAdd(zd + 7, v1.w * nrm);
}

// --- semantic attention: one wave per node ----------------------------------
__global__ __launch_bounds__(256)
void attention(const float* __restrict__ z, const float* __restrict__ W1,
               const float* __restrict__ b1, const float* __restrict__ w2,
               float* __restrict__ out, int n) {
  __shared__ float4 zs[8][NPATH][64];            // 32 KB
  int w = threadIdx.x >> 5;
  int lane = threadIdx.x & 31;
  int node = blockIdx.x * 8 + w;
  bool valid = node < n;

  if (valid) {
    for (int p = 0; p < NPATH; ++p) {
      const float4* zp = reinterpret_cast<const float4*>(z + ((size_t)p * n + node) * DDIM);
      zs[w][p][lane]      = zp[lane];
      zs[w][p][lane + 32] = zp[lane + 32];
    }
  }
  __syncthreads();

  if (valid) {
    float s[NPATH];
    float4 w2v = *reinterpret_cast<const float4*>(w2 + lane * 4);
    float4 b1v = *reinterpret_cast<const float4*>(b1 + lane * 4);
    for (int p = 0; p < NPATH; ++p) {
      float h0 = b1v.x, h1 = b1v.y, h2 = b1v.z, h3 = b1v.w;
      const float* zrow = reinterpret_cast<const float*>(&zs[w][p][0]);
      for (int k = 0; k < DDIM; ++k) {
        float zk = zrow[k];
        float4 wv = *reinterpret_cast<const float4*>(W1 + (size_t)k * DHID + lane * 4);
        h0 += zk * wv.x; h1 += zk * wv.y; h2 += zk * wv.z; h3 += zk * wv.w;
      }
      float sj = tanhf(h0) * w2v.x + tanhf(h1) * w2v.y +
                 tanhf(h2) * w2v.z + tanhf(h3) * w2v.w;
      for (int off = 16; off > 0; off >>= 1) sj += __shfl_xor(sj, off, 32);
      s[p] = sj;                                 // all lanes hold the full sum
    }
    float mx = fmaxf(fmaxf(s[0], s[1]), fmaxf(s[2], s[3]));
    float beta[NPATH], denom = 0.f;
    for (int p = 0; p < NPATH; ++p) { beta[p] = __expf(s[p] - mx); denom += beta[p]; }
    float inv = 1.0f / denom;
    float* op = out + (size_t)node * DDIM;
    for (int i = lane; i < 64; i += 32) {
      float4 acc = {0.f, 0.f, 0.f, 0.f};
      for (int p = 0; p < NPATH; ++p) {
        float bp = beta[p] * inv;
        float4 zv = zs[w][p][i];
        acc.x += bp * zv.x; acc.y += bp * zv.y;
        acc.z += bp * zv.z; acc.w += bp * zv.w;
      }
      reinterpret_cast<float4*>(op)[i] = acc;
    }
  }
}

// ---------------------------------------------------------------------------
extern "C" void kernel_launch(void* const* d_in, const int* in_sizes, int n_in,
                              void* d_out, int out_size, void* d_ws, size_t ws_size,
                              hipStream_t stream) {
  const float* x  = (const float*)d_in[0];
  const int*   ei = (const int*)d_in[1];
  const float* Ws = (const float*)d_in[2];
  const float* bs = (const float*)d_in[3];
  const float* W1 = (const float*)d_in[4];
  const float* b1 = (const float*)d_in[5];
  const float* w2 = (const float*)d_in[6];
  float* out = (float*)d_out;

  int n  = in_sizes[0] / DDIM;              // 50000
  int nE = in_sizes[1] / (NPATH * 2);       // 800000

  char* ws = (char*)d_ws;
  size_t off = 0;
  auto take = [&](size_t bytes) -> void* {
    void* p = ws + off;
    off = (off + bytes + 255) & ~(size_t)255;
    return p;
  };
  uint16_t* xh  = (uint16_t*)take((size_t)n * DDIM * 2);
  uint16_t* wpk = (uint16_t*)take((size_t)NPATH * KTILES * NTILES * 32 * 16 * 2);
  float* xw     = (float*)take((size_t)NPATH * n * DDIM * 4);
  float* z      = (float*)take((size_t)NPATH * n * DDIM * 4);
  float* deg    = (float*)take((size_t)NPATH * n * 4);
  float* dinv   = (float*)take((size_t)NPATH * n * 4);

  { int tot = n * DDIM;
    convert_x<<<(tot + 255) / 256, 256, 0, stream>>>(x, xh, tot); }
  { int tot = NPATH * KTILES * NTILES * 32;
    pack_w<<<(tot + 255) / 256, 256, 0, stream>>>(Ws, wpk); }
  { int mt = (n + 15) / 16;
    int waves = NPATH * mt * 4;
    gemm_xw<<<(waves + 3) / 4, 128, 0, stream>>>(xh, wpk, xw, n, mt); }
  { int tot = NPATH * n;
    deg_init<<<(tot + 255) / 256, 256, 0, stream>>>(deg, tot); }
  { long long tot = (long long)NPATH * nE;
    deg_count<<<(unsigned)((tot + 255) / 256), 256, 0, stream>>>(ei, deg, n, nE); }
  { int tot = NPATH * n;
    deg_fin<<<(tot + 255) / 256, 256, 0, stream>>>(deg, dinv, tot); }
  { long long tot = (long long)NPATH * n * DDIM;
    z_init<<<(unsigned)((tot + 255) / 256), 256, 0, stream>>>(xw, dinv, bs, z, n); }
  { long long waves = (long long)NPATH * nE;
    edge_scatter<<<(unsigned)((waves + 7) / 8), 256, 0, stream>>>(ei, xw, dinv, z, n, nE); }
  { attention<<<(n + 7) / 8, 256, 0, stream>>>(z, W1, b1, w2, out, n); }
}
